// SS2D_41394894799413
// MI455X (gfx1250) — matrix operand order repaired and checked by reading, not compiled, not run on hardware
//
#include <hip/hip_runtime.h>
#include <math.h>

typedef __attribute__((ext_vector_type(16))) __bf16 v16bf;
typedef __attribute__((ext_vector_type(8)))  float  v8f;

#define BB     2
#define DD     192
#define HH     64
#define LL     4096
#define KK     4
#define NN     16
#define RR     6
#define CPROJ  38
#define NCHUNK 64
#define CLEN   64

// Position in row-major x (h*W+w) that direction-k scan index l reads.
__device__ __forceinline__ int pos_for(int k, int l) {
  int ll = (k >= 2) ? (LL - 1 - l) : l;
  if (k & 1) { int h = ll & (HH - 1); int w = ll >> 6; return (h << 6) + w; }
  return ll;
}

// ---------------------------------------------------------------------------
// Kernel 0: LDS-tiled transpose x(B,D,L) -> xT(B,L,D). Both sides coalesced.
// ---------------------------------------------------------------------------
__global__ __launch_bounds__(256) void transpose_kernel(
    const float* __restrict__ x, float* __restrict__ xT) {
  __shared__ float s[32][33];
  int tile = blockIdx.x;                       // B * (D/32) * (L/32)
  int lt = tile % (LL / 32); int rest = tile / (LL / 32);
  int dt = rest % (DD / 32); int b = rest / (DD / 32);
  int tx = threadIdx.x & 31;
  int ty = threadIdx.x >> 5;                   // 0..7
  int l0 = lt * 32, d0 = dt * 32;
#pragma unroll
  for (int j = 0; j < 4; ++j) {
    int d = d0 + ty + j * 8;
    s[ty + j * 8][tx] = x[((size_t)b * DD + d) * LL + l0 + tx];
  }
  __syncthreads();
#pragma unroll
  for (int j = 0; j < 4; ++j) {
    int l = l0 + ty + j * 8;
    xT[((size_t)b * LL + l) * DD + d0 + tx] = s[tx][ty + j * 8];
  }
}

// ---------------------------------------------------------------------------
// Kernel 1: x_dbl[b,k,c,l] = sum_d W[k,c,d] * xs[b,k,d,l]   (c < 38)
// WMMA: M = c (3 tiles of 16, masked at 38), N = 16 l-columns, K-loop over D.
// B-matrix elements are consecutive d in xT -> vectorizable b128 loads.
// ---------------------------------------------------------------------------
__global__ __launch_bounds__(128) void proj_wmma_kernel(
    const float* __restrict__ xT, const float* __restrict__ Wp,
    float* __restrict__ xdbl) {
  int wave  = (blockIdx.x * blockDim.x + threadIdx.x) >> 5;   // 2048 waves
  int lane  = threadIdx.x & 31;
  int ltile = wave & 255;
  int bk    = wave >> 8;            // 0..7
  int k     = bk & 3, b = bk >> 2;
  int m     = lane & 15;            // A row within tile / B,C column
  int hs    = lane >> 4;

  int lcol = ltile * 16 + m;
  int p    = pos_for(k, lcol);
  const float* xrowT = xT + ((size_t)b * LL + p) * DD;  // contiguous in d
  const float* wk    = Wp + (size_t)k * CPROJ * DD;

  v8f acc0 = {}; v8f acc1 = {}; v8f acc2 = {};
  for (int kt = 0; kt < DD / 32; ++kt) {
    int dbase = kt * 32;
    v16bf bm;
#pragma unroll
    for (int e = 0; e < 16; ++e) {            // B: K = hs*16 + e (consecutive d)
      bm[e] = (__bf16)xrowT[dbase + hs * 16 + e];
    }
    v16bf am0, am1, am2;
#pragma unroll
    for (int e = 0; e < 16; ++e) {            // A: K = hs*16 + e
      int dIdx = dbase + hs * 16 + e;
      int c2  = 32 + m;
      int c2c = (c2 < CPROJ) ? c2 : (CPROJ - 1);  // clamped, branch-free
      am0[e] = (__bf16)wk[(0 + m)  * DD + dIdx];
      am1[e] = (__bf16)wk[(16 + m) * DD + dIdx];
      float f2 = wk[c2c * DD + dIdx];
      f2 = (c2 < CPROJ) ? f2 : 0.0f;
      am2[e] = (__bf16)f2;
    }
    acc0 = __builtin_amdgcn_wmma_f32_16x16x32_bf16(false, am0, false, bm, (short)0, acc0, false, false);
    acc1 = __builtin_amdgcn_wmma_f32_16x16x32_bf16(false, am1, false, bm, (short)0, acc1, false, false);
    acc2 = __builtin_amdgcn_wmma_f32_16x16x32_bf16(false, am2, false, bm, (short)0, acc2, false, false);
  }
  // C/D layout: vgpr r, lane-half hs -> row c = tile*16 + hs*8 + r ; column = m
  float* out = xdbl + (size_t)bk * CPROJ * LL + (ltile * 16 + m);
#pragma unroll
  for (int r = 0; r < 8; ++r) {
    int c0 = hs * 8 + r;
    out[(size_t)c0 * LL]        = acc0[r];
    out[(size_t)(16 + c0) * LL] = acc1[r];
    int c2 = 32 + c0;
    if (c2 < CPROJ) out[(size_t)c2 * LL] = acc2[r];
  }
}

// ---------------------------------------------------------------------------
// Kernel 2: delta[b,k,l,d] = softplus( sum_r dtW[k,d,r]*x_dbl[b,k,r,l] + bias )
// WMMA with K padded 6 -> 32 (branch-free zero padding). Output is l-major so
// the scan phases read it coalesced (lanes = consecutive d).
// ---------------------------------------------------------------------------
__global__ __launch_bounds__(256) void dt_wmma_kernel(
    const float* __restrict__ xdbl, const float* __restrict__ dtW,
    const float* __restrict__ dtb, float* __restrict__ delta) {
  int wave  = (blockIdx.x * blockDim.x + threadIdx.x) >> 5;   // 24576 waves
  int lane  = threadIdx.x & 31;
  int ltile = wave % 256;
  int rest  = wave / 256;
  int dtile = rest % (DD / 16);
  int bk    = rest / (DD / 16);
  int k     = bk & 3;
  int m     = lane & 15;
  int hs    = lane >> 4;
  int lcol  = ltile * 16 + m;

  v16bf am, bm;
#pragma unroll
  for (int e = 0; e < 16; ++e) {
    int kkA  = hs * 16 + e;
    int kkAc = (kkA < RR) ? kkA : (RR - 1);
    int dch  = dtile * 16 + m;
    float fa = dtW[((size_t)k * DD + dch) * RR + kkAc];
    fa = (kkA < RR) ? fa : 0.0f;
    am[e] = (__bf16)fa;
    int kkB  = hs * 16 + e;
    int kkBc = (kkB < RR) ? kkB : (RR - 1);
    float fb = xdbl[((size_t)bk * CPROJ + kkBc) * LL + lcol];
    fb = (kkB < RR) ? fb : 0.0f;
    bm[e] = (__bf16)fb;
  }
  v8f acc = {};
  acc = __builtin_amdgcn_wmma_f32_16x16x32_bf16(false, am, false, bm, (short)0, acc, false, false);
#pragma unroll
  for (int r = 0; r < 8; ++r) {
    int dch = dtile * 16 + hs * 8 + r;
    float v = acc[r] + dtb[k * DD + dch];
    float sp = (v > 20.0f) ? v : log1pf(__expf(v));
    delta[((size_t)bk * LL + lcol) * DD + dch] = sp;   // (b,k,l,d)
  }
}

// ---------------------------------------------------------------------------
// Kernel 3: per-chunk local scan -> transition (P, q). One thread per
// (b,k,chunk,d); lanes = consecutive d -> delta/u loads are one 128B line per
// wave per step; Bs column loads are uniform-address broadcasts.
// ---------------------------------------------------------------------------
__global__ __launch_bounds__(256) void scan_chunks_kernel(
    const float* __restrict__ xT, const float* __restrict__ xdbl,
    const float* __restrict__ delta, const float* __restrict__ A_logs,
    float* __restrict__ Pq) {
  int t = blockIdx.x * blockDim.x + threadIdx.x;   // 98304
  if (t >= BB * KK * NCHUNK * DD) return;
  int d = t % DD; int rest = t / DD;
  int chunk = rest & (NCHUNK - 1); int bk = rest >> 6;
  int k = bk & 3, b = bk >> 2;

  float Av[NN];
#pragma unroll
  for (int n = 0; n < NN; ++n) Av[n] = -__expf(A_logs[((size_t)k * DD + d) * NN + n]);

  const float* del   = delta + (size_t)bk * LL * DD + d;     // + l*DD
  const float* Bbase = xdbl + ((size_t)bk * CPROJ + RR) * LL;
  const float* xTb   = xT + (size_t)b * LL * DD + d;         // + pos*DD

  float h[NN], P[NN];
#pragma unroll
  for (int n = 0; n < NN; ++n) { h[n] = 0.0f; P[n] = 1.0f; }

  int l0 = chunk * CLEN;
  for (int i = 0; i < CLEN; ++i) {
    int l = l0 + i;
    int lpf = (l + 16 < LL) ? (l + 16) : (LL - 1);
    __builtin_prefetch(&del[(size_t)lpf * DD], 0, 1);        // global_prefetch_b8
    float dl = del[(size_t)l * DD];
    float u  = xTb[(size_t)pos_for(k, l) * DD];
    float du = dl * u;
#pragma unroll
    for (int n = 0; n < NN; ++n) {
      float a = __expf(dl * Av[n]);
      h[n] = a * h[n] + du * Bbase[(size_t)n * LL + l];
      P[n] *= a;
    }
  }
  float* out = Pq + (size_t)t * (2 * NN);
#pragma unroll
  for (int n = 0; n < NN; ++n) { out[n] = P[n]; out[NN + n] = h[n]; }
}

// ---------------------------------------------------------------------------
// Kernel 4: serial combine of chunk transitions -> h_in for each chunk.
// ---------------------------------------------------------------------------
__global__ __launch_bounds__(256) void combine_kernel(
    const float* __restrict__ Pq, float* __restrict__ hin) {
  int t = blockIdx.x * blockDim.x + threadIdx.x;   // 1536
  if (t >= BB * KK * DD) return;
  int d = t % DD; int bk = t / DD;
  float h[NN];
#pragma unroll
  for (int n = 0; n < NN; ++n) h[n] = 0.0f;
  for (int c = 0; c < NCHUNK; ++c) {
    size_t idx = ((size_t)bk * NCHUNK + c) * DD + d;
    float* ho = hin + idx * NN;
    const float* pq = Pq + idx * (2 * NN);
#pragma unroll
    for (int n = 0; n < NN; ++n) ho[n] = h[n];
#pragma unroll
    for (int n = 0; n < NN; ++n) h[n] = pq[n] * h[n] + pq[NN + n];
  }
}

// ---------------------------------------------------------------------------
// Kernel 5: re-scan chunks from correct h_in; y = C.h + Ds*u.
// y written l-major (b,k,l,d) so stores and the merge reads are coalesced.
// ---------------------------------------------------------------------------
__global__ __launch_bounds__(256) void scan_final_kernel(
    const float* __restrict__ xT, const float* __restrict__ xdbl,
    const float* __restrict__ delta, const float* __restrict__ A_logs,
    const float* __restrict__ Ds, const float* __restrict__ hin,
    float* __restrict__ y) {
  int t = blockIdx.x * blockDim.x + threadIdx.x;   // 98304
  if (t >= BB * KK * NCHUNK * DD) return;
  int d = t % DD; int rest = t / DD;
  int chunk = rest & (NCHUNK - 1); int bk = rest >> 6;
  int k = bk & 3, b = bk >> 2;

  float Av[NN], h[NN];
#pragma unroll
  for (int n = 0; n < NN; ++n) {
    Av[n] = -__expf(A_logs[((size_t)k * DD + d) * NN + n]);
    h[n]  = hin[(size_t)t * NN + n];
  }
  float Dval = Ds[k * DD + d];

  const float* del   = delta + (size_t)bk * LL * DD + d;
  const float* Bbase = xdbl + ((size_t)bk * CPROJ + RR) * LL;
  const float* Cbase = xdbl + ((size_t)bk * CPROJ + RR + NN) * LL;
  const float* xTb   = xT + (size_t)b * LL * DD + d;
  float* yout        = y + (size_t)bk * LL * DD + d;         // + l*DD

  int l0 = chunk * CLEN;
  for (int i = 0; i < CLEN; ++i) {
    int l = l0 + i;
    int lpf = (l + 16 < LL) ? (l + 16) : (LL - 1);
    __builtin_prefetch(&del[(size_t)lpf * DD], 0, 1);
    float dl = del[(size_t)l * DD];
    float u  = xTb[(size_t)pos_for(k, l) * DD];
    float du = dl * u;
    float acc = 0.0f;
#pragma unroll
    for (int n = 0; n < NN; ++n) {
      float a = __expf(dl * Av[n]);
      h[n] = a * h[n] + du * Bbase[(size_t)n * LL + l];
      acc += Cbase[(size_t)n * LL + l] * h[n];
    }
    yout[(size_t)l * DD] = acc + Dval * u;
  }
}

// ---------------------------------------------------------------------------
// Kernel 6: CrossMerge: undo flips/transpose, sum 4 directions (all coalesced
// in d with the l-major y layout). ym is (b,l,d).
// ---------------------------------------------------------------------------
__global__ __launch_bounds__(256) void merge_kernel(
    const float* __restrict__ y, float* __restrict__ ym) {
  int t = blockIdx.x * blockDim.x + threadIdx.x;   // B*L*D, d fastest
  if (t >= BB * LL * DD) return;
  int d = t % DD; int rest = t / DD;
  int l = rest % LL; int b = rest / LL;
  int h = l >> 6, w = l & (HH - 1);
  int p1 = (w << 6) + h;                 // w*H + h
  size_t base = (size_t)b * KK * LL * DD;
  float v0 = y[base + (0 * (size_t)LL + l) * DD + d];
  float v2 = y[base + (2 * (size_t)LL + (LL - 1 - l)) * DD + d];
  float v1 = y[base + (1 * (size_t)LL + p1) * DD + d];
  float v3 = y[base + (3 * (size_t)LL + (LL - 1 - p1)) * DD + d];
  ym[((size_t)b * LL + l) * DD + d] = v0 + v1 + v2 + v3;
}

// ---------------------------------------------------------------------------
// Kernel 7: LayerNorm over channels; reads a contiguous 192-float row per
// pixel; output (B,H,W,D).
// ---------------------------------------------------------------------------
__global__ __launch_bounds__(192) void ln_kernel(
    const float* __restrict__ ym, const float* __restrict__ lw,
    const float* __restrict__ lb, float* __restrict__ out) {
  int bl = blockIdx.x;            // B*L blocks
  int l = bl % LL; int b = bl / LL;
  int d = threadIdx.x;            // 0..191
  float v = ym[((size_t)b * LL + l) * DD + d];
  __shared__ float s1[DD], s2[DD], mv[2];
  s1[d] = v; s2[d] = v * v;
  __syncthreads();
  for (int s = 96; s >= 3; s >>= 1) {
    if (d < s) { s1[d] += s1[d + s]; s2[d] += s2[d + s]; }
    __syncthreads();
  }
  if (d == 0) {
    float sum = s1[0] + s1[1] + s1[2];
    float sq  = s2[0] + s2[1] + s2[2];
    float mu  = sum / (float)DD;
    float var = sq / (float)DD - mu * mu;
    mv[0] = mu; mv[1] = rsqrtf(var + 1e-5f);
  }
  __syncthreads();
  out[((size_t)b * LL + l) * DD + d] = (v - mv[0]) * mv[1] * lw[d] + lb[d];
}

extern "C" void kernel_launch(void* const* d_in, const int* in_sizes, int n_in,
                              void* d_out, int out_size, void* d_ws, size_t ws_size,
                              hipStream_t stream) {
  const float* x    = (const float*)d_in[0];
  const float* xpw  = (const float*)d_in[1];
  const float* dtW  = (const float*)d_in[2];
  const float* dtb  = (const float*)d_in[3];
  const float* Alog = (const float*)d_in[4];
  const float* Ds   = (const float*)d_in[5];
  const float* lnw  = (const float*)d_in[6];
  const float* lnb  = (const float*)d_in[7];
  float* out = (float*)d_out;

  float* ws    = (float*)d_ws;
  float* xT    = ws;                                          // B*L*D
  float* xdbl  = xT    + (size_t)BB * LL * DD;                // B*K*38*L
  float* delta = xdbl  + (size_t)BB * KK * CPROJ * LL;        // B*K*L*D
  float* Pq    = delta + (size_t)BB * KK * LL * DD;           // 98304*32
  float* hin   = Pq    + (size_t)BB * KK * NCHUNK * DD * 32;  // 98304*16
  float* ybuf  = hin   + (size_t)BB * KK * NCHUNK * DD * 16;  // B*K*L*D
  float* ym    = ybuf  + (size_t)BB * KK * LL * DD;           // B*L*D

  // 0) transpose x -> xT (d-fastest for coalesced scan-phase reads)
  transpose_kernel<<<BB * (DD / 32) * (LL / 32), 256, 0, stream>>>(x, xT);
  // 1) x projection GEMM (WMMA bf16): 2048 waves, 4 waves/block
  proj_wmma_kernel<<<512, 128, 0, stream>>>(xT, xpw, xdbl);
  // 2) dt projection GEMM (WMMA bf16) + bias + softplus: 24576 waves
  dt_wmma_kernel<<<3072, 256, 0, stream>>>(xdbl, dtW, dtb, delta);
  // 3) chunked scan phase A
  scan_chunks_kernel<<<(BB * KK * NCHUNK * DD + 255) / 256, 256, 0, stream>>>(
      xT, xdbl, delta, Alog, Pq);
  // 4) chunk combine
  combine_kernel<<<(BB * KK * DD + 255) / 256, 256, 0, stream>>>(Pq, hin);
  // 5) chunked scan phase C
  scan_final_kernel<<<(BB * KK * NCHUNK * DD + 255) / 256, 256, 0, stream>>>(
      xT, xdbl, delta, Alog, Ds, hin, ybuf);
  // 6) CrossMerge
  merge_kernel<<<(BB * LL * DD + 255) / 256, 256, 0, stream>>>(ybuf, ym);
  // 7) LayerNorm -> (B,H,W,D)
  ln_kernel<<<BB * LL, DD, 0, stream>>>(ym, lnw, lnb, out);
}
